// WMHCA_Block_42571715838264
// MI455X (gfx1250) — matrix operand order repaired and checked by reading, not compiled, not run on hardware
//
#include <hip/hip_runtime.h>

#define BATCH 128
#define CCH   512
#define CRR   256
#define HWN   1024   // 32*32 spatial
#define TOK   256    // 16*16 tokens per window

typedef __attribute__((ext_vector_type(16))) __bf16 v16bf;
typedef __attribute__((ext_vector_type(8)))  float  v8f;
typedef __attribute__((ext_vector_type(4)))  unsigned int u32x4;
typedef __attribute__((ext_vector_type(8)))  int i32x8;
typedef __attribute__((ext_vector_type(4)))  int i32x4;

union Frag16 {
  v16bf v;
  unsigned short u[16];
  unsigned int d[8];
  uint4 q[2];
};

// pack two f32 -> two bf16 in one dword (round-half bias + v_perm_b32)
__device__ __forceinline__ unsigned int pkbf(float lo, float hi) {
  unsigned int a = __builtin_bit_cast(unsigned int, lo) + 0x8000u;
  unsigned int b = __builtin_bit_cast(unsigned int, hi) + 0x8000u;
  return __builtin_amdgcn_perm(b, a, 0x07060302u);
}
__device__ __forceinline__ unsigned short sbf(float f) {
  return (unsigned short)((__builtin_bit_cast(unsigned int, f) + 0x8000u) >> 16);
}

// A-fragment (16x32 bf16): lane holds 16 contiguous elems
__device__ __forceinline__ void loadA(Frag16& f, const unsigned short* base) {
  const uint4* p = (const uint4*)base;
  f.q[0] = p[0];
  f.q[1] = p[1];   // +8 ushorts
}
// B-fragment (32x16 bf16): lane holds 16 contiguous elems
__device__ __forceinline__ void loadB(Frag16& f, const unsigned short* base) {
  const uint4* p = (const uint4*)base;
  f.q[0] = p[0];
  f.q[1] = p[1];
}

// window w (0..3), token t (0..255) -> flat 32x32 spatial index
__device__ __forceinline__ int hw_of(int w, int t) {
  return (((w >> 1) * 16 + (t >> 4)) << 5) + ((w & 1) * 16) + (t & 15);
}
// flat spatial -> (w, t)
__device__ __forceinline__ void wt_of(int hw, int& w, int& t) {
  const int y = hw >> 5, x = hw & 31;
  w = ((y >> 4) << 1) | (x >> 4);
  t = ((y & 15) << 4) | (x & 15);
}

__device__ __forceinline__ v8f wmma_bf16(const Frag16& a, const Frag16& b, v8f c) {
  return __builtin_amdgcn_wmma_f32_16x16x32_bf16(false, a.v, false, b.v, (short)0, c,
                                                 false, false);
}

#if __has_builtin(__builtin_amdgcn_tensor_load_to_lds)
#define USE_TDM 1
// TDM load of a (up to 3D) bf16 tile into LDS with pitch-40 padding:
//   X = tile0 contiguous elems, Y = tile1 rows (stride0 elems), Z = tile2 (stride1)
//   pad 4 dwords after every 16 dwords stored -> 32-elem rows at pitch 40 ushorts
__device__ __forceinline__ void tdm_load_bf16(
    unsigned lds_addr, const void* gaddr,
    unsigned tile0, unsigned tile1, unsigned tile2,
    unsigned long long stride0, unsigned long long stride1) {
  const unsigned long long ga = (unsigned long long)(size_t)gaddr;
  const unsigned big = 1u << 20;   // generous tensor dims (no OOB clipping)
  u32x4 g0;
  g0[0] = 1u;                                            // count=1 (valid D#)
  g0[1] = lds_addr;                                      // LDS byte address
  g0[2] = (unsigned)ga;                                  // global_addr[31:0]
  g0[3] = ((unsigned)(ga >> 32) & 0x01FFFFFFu) | 0x80000000u; // addr[56:32] | type=2
  i32x8 g1;
  g1[0] = (int)((1u << 16)      // data_size = 2 bytes
              | (1u << 20)      // pad_enable
              | (3u << 22)      // pad_interval: 16 dwords
              | (3u << 25));    // pad_amount: 4 dwords
  g1[1] = (int)((big & 0xFFFFu) << 16);                              // dim0 lo16
  g1[2] = (int)(((big >> 16) & 0xFFFFu) | ((big & 0xFFFFu) << 16));  // dim0 hi | dim1 lo
  g1[3] = (int)(((big >> 16) & 0xFFFFu) | (tile0 << 16));            // dim1 hi | tile0
  g1[4] = (int)(tile1 | (tile2 << 16));                              // tile1 | tile2
  g1[5] = (int)(unsigned)stride0;                                    // stride0[31:0]
  g1[6] = (int)(((unsigned)(stride0 >> 32) & 0xFFFFu) |
                ((unsigned)stride1 << 16));                          // s0 hi | s1 lo
  g1[7] = (int)(unsigned)(stride1 >> 16);                            // stride1[47:16]
  i32x4 g2;
  g2[0] = (int)big; g2[1] = (int)big; g2[2] = 0; g2[3] = 0;          // dim2/dim3
  i32x4 g3;
  g3[0] = 0; g3[1] = 0; g3[2] = 0; g3[3] = 0;
  i32x8 g4;                                                          // extended words
  g4[0] = 0; g4[1] = 0; g4[2] = 0; g4[3] = 0;
  g4[4] = 0; g4[5] = 0; g4[6] = 0; g4[7] = 0;
  // 6-arg (clang-23 / therock) form: groups 0-3, extended group, cpol
  __builtin_amdgcn_tensor_load_to_lds(g0, g1, g2, g3, g4, 0);
}
#else
#define USE_TDM 0
#endif

// ---------------------------------------------------------------------------
// 0) one-shot fp32 -> bf16 conversion of the four weight matrices (131072 each)
// ---------------------------------------------------------------------------
__global__ __launch_bounds__(256)
void convw_kernel(const float* __restrict__ s0, const float* __restrict__ s1,
                  const float* __restrict__ s2, const float* __restrict__ s3,
                  unsigned short* __restrict__ dst) {
  const float* src = (blockIdx.y == 0) ? s0 : (blockIdx.y == 1) ? s1
                    : (blockIdx.y == 2) ? s2 : s3;
  const int i = (blockIdx.x * 256 + threadIdx.x) * 2;
  unsigned int* d = (unsigned int*)(dst + (size_t)blockIdx.y * 131072);
  d[i >> 1] = pkbf(src[i], src[i + 1]);
}

// ---------------------------------------------------------------------------
// 1) phi = Wp*xa, theta = Wt*xb, g = Wg*(xa+xb)   (per-batch GEMM, K=512)
//    phi/theta emitted TOKEN-MAJOR [b,w,t,cr] via per-wave LDS transpose;
//    g emitted [b,cr,hw].
// ---------------------------------------------------------------------------
__global__ __launch_bounds__(256)
void proj_kernel(const float* __restrict__ xa, const float* __restrict__ xb,
                 const unsigned short* __restrict__ wpb,
                 const unsigned short* __restrict__ wtb,
                 const unsigned short* __restrict__ wgb,
                 unsigned short* __restrict__ phiT, unsigned short* __restrict__ thT,
                 unsigned short* __restrict__ gg) {
  __shared__ unsigned short tr[8 * 32 * 40];   // per-wave 32 rows x pitch 40
  const int b    = blockIdx.y;
  const int hw0  = blockIdx.x * 32;
  const int wave = threadIdx.x >> 5, lane = threadIdx.x & 31;
  const int ln = lane & 15, kh = lane >> 4;
  const int row0 = wave * 32;
  const size_t xoff = (size_t)b * CCH * HWN;

  v8f ap[2][2] = {}, at[2][2] = {}, ag[2][2] = {};

  for (int kc = 0; kc < CCH; kc += 32) {
    Frag16 fp[2], ft[2], fg[2];
#pragma unroll
    for (int rt = 0; rt < 2; ++rt) {
      const size_t ro = (size_t)(row0 + rt * 16 + ln) * CCH + kc + kh * 16;
      loadA(fp[rt], wpb + ro);
      loadA(ft[rt], wtb + ro);
      loadA(fg[rt], wgb + ro);
    }
#pragma unroll
    for (int ct = 0; ct < 2; ++ct) {
      Frag16 ba, bb, bg;
      const size_t base = xoff + (size_t)(kc + kh * 16) * HWN + hw0 + ct * 16 + ln;
#pragma unroll
      for (int p = 0; p < 8; ++p) {
        const size_t a0 = base + (size_t)(2 * p) * HWN;
        const float fa0 = xa[a0], fa1 = xa[a0 + HWN];
        const float fb0 = xb[a0], fb1 = xb[a0 + HWN];
        ba.d[p] = pkbf(fa0, fa1);
        bb.d[p] = pkbf(fb0, fb1);
        bg.d[p] = pkbf(fa0 + fb0, fa1 + fb1);
      }
#pragma unroll
      for (int rt = 0; rt < 2; ++rt) {
        ap[rt][ct] = wmma_bf16(fp[rt], ba, ap[rt][ct]);
        at[rt][ct] = wmma_bf16(ft[rt], bb, at[rt][ct]);
        ag[rt][ct] = wmma_bf16(fg[rt], bg, ag[rt][ct]);
      }
    }
  }

  // ---- g: direct [cr,hw] stores ----
  const size_t ooff = (size_t)b * CRR * HWN;
#pragma unroll
  for (int rt = 0; rt < 2; ++rt)
#pragma unroll
    for (int ct = 0; ct < 2; ++ct)
#pragma unroll
      for (int r = 0; r < 8; ++r) {
        const int o = row0 + rt * 16 + r + (kh ? 8 : 0);
        gg[ooff + (size_t)o * HWN + hw0 + ct * 16 + ln] = sbf(ag[rt][ct][r]);
      }

  // ---- phi/theta: per-wave LDS transpose, then token-major rows ----
  unsigned short* ltr = tr + wave * 32 * 40;
  int wdw, tok;
  wt_of(hw0 + lane, wdw, tok);
  const size_t trow = ((size_t)(b * 4 + wdw) * TOK + tok) * CRR + row0;

#pragma unroll
  for (int mat = 0; mat < 2; ++mat) {
#pragma unroll
    for (int rt = 0; rt < 2; ++rt)
#pragma unroll
      for (int ct = 0; ct < 2; ++ct) {
        const v8f& a = mat ? at[rt][ct] : ap[rt][ct];
        uint4 vv;
        vv.x = pkbf(a[0], a[1]);
        vv.y = pkbf(a[2], a[3]);
        vv.z = pkbf(a[4], a[5]);
        vv.w = pkbf(a[6], a[7]);
        *(uint4*)(ltr + (ct * 16 + ln) * 40 + rt * 16 + kh * 8) = vv;
      }
    // wave-local read-back (DS ops are in-order within a wave)
    const uint4* sp = (const uint4*)(ltr + lane * 40);
    uint4* dp = (uint4*)((mat ? thT : phiT) + trow);
#pragma unroll
    for (int i = 0; i < 4; ++i) dp[i] = sp[i];
  }
}

// ---------------------------------------------------------------------------
// 2) attn[b,w] = theta_w^T @ phi_w  (256x256x256), + tile sums for SE
// ---------------------------------------------------------------------------
__global__ __launch_bounds__(256)
void attn_kernel(const unsigned short* __restrict__ phiT,
                 const unsigned short* __restrict__ thT,
                 float* __restrict__ attn, float* __restrict__ sums) {
  const int bw = blockIdx.y;
  const int t0 = (blockIdx.x >> 1) * 128 + ((threadIdx.x >> 5) >> 1) * 32;
  const int s0 = (blockIdx.x & 1) * 128 + ((threadIdx.x >> 5) & 1) * 64;
  const int lane = threadIdx.x & 31, ln = lane & 15, kh = lane >> 4;
  const size_t woff = (size_t)bw * TOK * CRR;

  v8f acc[2][4] = {};
  for (int kc = 0; kc < CRR; kc += 32) {
    Frag16 A[2], Bf[4];
#pragma unroll
    for (int rt = 0; rt < 2; ++rt)
      loadA(A[rt], thT + woff + (size_t)(t0 + rt * 16 + ln) * CRR + kc + kh * 16);
#pragma unroll
    for (int ct = 0; ct < 4; ++ct)
      loadB(Bf[ct], phiT + woff + (size_t)(s0 + ct * 16 + ln) * CRR + kc + kh * 16);
#pragma unroll
    for (int rt = 0; rt < 2; ++rt)
#pragma unroll
      for (int ct = 0; ct < 4; ++ct)
        acc[rt][ct] = wmma_bf16(A[rt], Bf[ct], acc[rt][ct]);
  }
  float lsum = 0.f;
  const size_t aoff = (size_t)bw * TOK * TOK;
#pragma unroll
  for (int rt = 0; rt < 2; ++rt)
#pragma unroll
    for (int ct = 0; ct < 4; ++ct)
#pragma unroll
      for (int r = 0; r < 8; ++r) {
        const int t = t0 + rt * 16 + r + (kh ? 8 : 0);
        const int s = s0 + ct * 16 + ln;
        const float v = acc[rt][ct][r];
        attn[aoff + (size_t)t * TOK + s] = v;
        lsum += v;
      }
  for (int off = 16; off > 0; off >>= 1) lsum += __shfl_xor(lsum, off, 32);
  if (lane == 0) atomicAdd(&sums[bw], lsum);
}

// ---------------------------------------------------------------------------
// 3) SE MLP: y = sigmoid(relu(mean @ w1^T) @ w2^T)
// ---------------------------------------------------------------------------
__global__ void se_kernel(const float* __restrict__ sums,
                          const float* __restrict__ w1, const float* __restrict__ w2,
                          float* __restrict__ y) {
  const int b = threadIdx.x;
  if (b >= BATCH) return;
  float m[4], h[2];
#pragma unroll
  for (int i = 0; i < 4; ++i) m[i] = sums[b * 4 + i] * (1.0f / 65536.0f);
#pragma unroll
  for (int r = 0; r < 2; ++r) {
    float s = 0.f;
#pragma unroll
    for (int j = 0; j < 4; ++j) s += w1[r * 4 + j] * m[j];
    h[r] = s > 0.f ? s : 0.f;
  }
#pragma unroll
  for (int i = 0; i < 4; ++i) {
    float s = 0.f;
#pragma unroll
    for (int k = 0; k < 2; ++k) s += w2[i * 2 + k] * h[k];
    y[b * 4 + i] = 1.0f / (1.0f + __expf(-s));
  }
}

// ---------------------------------------------------------------------------
// 4) SE scale + softmax over the 4 windows, emit bf16 probabilities
// ---------------------------------------------------------------------------
__global__ __launch_bounds__(256)
void softmax_kernel(const float* __restrict__ attn, const float* __restrict__ y,
                    unsigned short* __restrict__ attnh) {
  const size_t idx = (size_t)blockIdx.x * 256 + threadIdx.x;
  const int b = (int)(idx >> 16);
  const int ts = (int)(idx & 65535);
  float a[4], mx = -3.0e38f;
#pragma unroll
  for (int w = 0; w < 4; ++w) {
    a[w] = attn[((size_t)(b * 4 + w) << 16) + ts] * y[b * 4 + w];
    mx = fmaxf(mx, a[w]);
  }
  float s = 0.f;
#pragma unroll
  for (int w = 0; w < 4; ++w) { a[w] = __expf(a[w] - mx); s += a[w]; }
  const float inv = 1.0f / s;
#pragma unroll
  for (int w = 0; w < 4; ++w)
    attnh[((size_t)(b * 4 + w) << 16) + ts] = sbf(a[w] * inv);
}

// ---------------------------------------------------------------------------
// 5) out1T[b,w,t,cr] = sum_s P[t,s] * g[cr,hw(s)]  computed as (M=cr, N=t).
//    Tiles staged by the Tensor Data Mover (double-buffered, TENSORcnt
//    pipelined); TDM pad feature reproduces the pitch-40 LDS layout.
//    Epilogue: block LDS transpose -> token-major out1.
// ---------------------------------------------------------------------------
__global__ __launch_bounds__(256)
void outw_kernel(const unsigned short* __restrict__ gg,
                 const unsigned short* __restrict__ attnh,
                 unsigned short* __restrict__ out1T) {
  __shared__ unsigned short smem[2 * 10240];   // 2 stages x (A 5120 + B 5120) = 40960B
  const int bw = blockIdx.y;  const int b = bw >> 2, w = bw & 3;
  const int cr0 = (blockIdx.x >> 1) * 128;
  const int t0b = (blockIdx.x & 1) * 128;
  const int wave = threadIdx.x >> 5, lane = threadIdx.x & 31;
  const int ln = lane & 15, kh = lane >> 4;
  const int rB = (wave >> 1) * 32;
  const int cB = (wave & 1) * 64;
  const size_t goff = (size_t)b * CRR * HWN;
  const size_t aoff = (size_t)bw * TOK * TOK;

#if USE_TDM
  // TDM issue of one K-stage (g tile: 3D 16x2x128; P tile: 2D 32x128)
  auto issue = [&](int kc, int buf) {
    unsigned short* la = smem + buf * 10240;
    tdm_load_bf16((unsigned)(size_t)la,
                  gg + goff + (size_t)cr0 * HWN + hw_of(w, kc),
                  16, 2, 128, 32ull, 1024ull);
    tdm_load_bf16((unsigned)(size_t)(la + 5120),
                  attnh + aoff + (size_t)t0b * TOK + kc,
                  32, 128, 0, 256ull, 0ull);
  };
  if (wave == 0) issue(0, 0);
#else
  const int li = threadIdx.x >> 1;
  const int lj = (threadIdx.x & 1) * 16;
#endif

  v8f acc[2][4] = {};
  for (int kc = 0; kc < TOK; kc += 32) {
    const int cur = (kc >> 5) & 1;
    unsigned short* ldsA = smem + cur * 10240;
    unsigned short* ldsB = ldsA + 5120;
#if USE_TDM
    if (wave == 0) {
      if (kc + 32 < TOK) {
        issue(kc + 32, cur ^ 1);
        __builtin_amdgcn_s_wait_tensorcnt(2);   // pair for stage kc is complete
      } else {
        __builtin_amdgcn_s_wait_tensorcnt(0);
      }
    }
    __syncthreads();
#else
    {
      const uint4* sa = (const uint4*)(gg + goff + (size_t)(cr0 + li) * HWN +
                                       hw_of(w, kc + lj));
      const uint4* sb = (const uint4*)(attnh + aoff + (size_t)(t0b + li) * TOK + kc + lj);
      uint4 a0 = sa[0], a1 = sa[1], b0 = sb[0], b1 = sb[1];
      uint4* da = (uint4*)(ldsA + li * 40 + lj);
      uint4* db = (uint4*)(ldsB + li * 40 + lj);
      da[0] = a0; da[1] = a1;
      db[0] = b0; db[1] = b1;
    }
    __syncthreads();
#endif
    Frag16 A[2], Bf[4];
#pragma unroll
    for (int rt = 0; rt < 2; ++rt)
      loadA(A[rt], ldsA + (rB + rt * 16 + ln) * 40 + kh * 16);
#pragma unroll
    for (int ct = 0; ct < 4; ++ct)
      loadB(Bf[ct], ldsB + (cB + ct * 16 + ln) * 40 + kh * 16);
#pragma unroll
    for (int rt = 0; rt < 2; ++rt)
#pragma unroll
      for (int ct = 0; ct < 4; ++ct)
        acc[rt][ct] = wmma_bf16(A[rt], Bf[ct], acc[rt][ct]);
    __syncthreads();
  }

  // transpose 128cr x 128t tile through LDS: smem[t][cr], pitch 136
#pragma unroll
  for (int rt = 0; rt < 2; ++rt)
#pragma unroll
    for (int ct = 0; ct < 4; ++ct) {
      const int n = cB + ct * 16 + ln;          // t_local
      const int m = rB + rt * 16 + kh * 8;      // cr_local base (8 consecutive)
      uint4 vv;
      vv.x = pkbf(acc[rt][ct][0], acc[rt][ct][1]);
      vv.y = pkbf(acc[rt][ct][2], acc[rt][ct][3]);
      vv.z = pkbf(acc[rt][ct][4], acc[rt][ct][5]);
      vv.w = pkbf(acc[rt][ct][6], acc[rt][ct][7]);
      *(uint4*)(smem + n * 136 + m) = vv;
    }
  __syncthreads();
  {
    const int t = threadIdx.x >> 1, half = threadIdx.x & 1;
    const uint4* sp = (const uint4*)(smem + t * 136 + half * 64);
    uint4* dp = (uint4*)(out1T + ((size_t)bw * TOK + t0b + t) * CRR + cr0 + half * 64);
#pragma unroll
    for (int i = 0; i < 8; ++i) dp[i] = sp[i];
  }
}

// ---------------------------------------------------------------------------
// 6) res = W_mask(512x256) @ fold(out1) + xa + xb   (fp32 out)
// ---------------------------------------------------------------------------
__global__ __launch_bounds__(256)
void mask_kernel(const unsigned short* __restrict__ wmb,
                 const unsigned short* __restrict__ out1T,
                 const float* __restrict__ xa, const float* __restrict__ xb,
                 float* __restrict__ out) {
  const int b = blockIdx.z;
  const int row0 = blockIdx.y * 256 + (threadIdx.x >> 5) * 32;
  const int hw0 = blockIdx.x * 32;
  const int lane = threadIdx.x & 31, ln = lane & 15, kh = lane >> 4;

  size_t brow[2];
#pragma unroll
  for (int ct = 0; ct < 2; ++ct) {
    int wdw, tok;
    wt_of(hw0 + ct * 16 + ln, wdw, tok);
    brow[ct] = ((size_t)(b * 4 + wdw) * TOK + tok) * CRR;
  }

  v8f acc[2][2] = {};
  for (int kc = 0; kc < CRR; kc += 32) {
    Frag16 A[2], Bf[2];
#pragma unroll
    for (int rt = 0; rt < 2; ++rt)
      loadA(A[rt], wmb + (size_t)(row0 + rt * 16 + ln) * CRR + kc + kh * 16);
#pragma unroll
    for (int ct = 0; ct < 2; ++ct)
      loadB(Bf[ct], out1T + brow[ct] + kc + kh * 16);
#pragma unroll
    for (int rt = 0; rt < 2; ++rt)
#pragma unroll
      for (int ct = 0; ct < 2; ++ct)
        acc[rt][ct] = wmma_bf16(A[rt], Bf[ct], acc[rt][ct]);
  }
  const size_t xoff = (size_t)b * CCH * HWN;
#pragma unroll
  for (int rt = 0; rt < 2; ++rt)
#pragma unroll
    for (int ct = 0; ct < 2; ++ct)
#pragma unroll
      for (int r = 0; r < 8; ++r) {
        const int o = row0 + rt * 16 + r + (kh ? 8 : 0);
        const size_t a = xoff + (size_t)o * HWN + hw0 + ct * 16 + ln;
        out[a] = acc[rt][ct][r] + xa[a] + xb[a];
      }
}

extern "C" void kernel_launch(void* const* d_in, const int* in_sizes, int n_in,
                              void* d_out, int out_size, void* d_ws, size_t ws_size,
                              hipStream_t stream) {
  (void)in_sizes; (void)n_in; (void)out_size; (void)ws_size;
  const float* xa = (const float*)d_in[0];
  const float* xb = (const float*)d_in[1];
  const float* Wp = (const float*)d_in[2];
  const float* Wt = (const float*)d_in[3];
  const float* Wg = (const float*)d_in[4];
  const float* Wm = (const float*)d_in[5];
  const float* w1 = (const float*)d_in[6];
  const float* w2 = (const float*)d_in[7];

  char* ws = (char*)d_ws;
  const size_t SZ = (size_t)BATCH * CRR * HWN * sizeof(unsigned short); // 67.1 MB
  unsigned short* phiT  = (unsigned short*)(ws);            // [b,w,t,cr]
  unsigned short* thT   = (unsigned short*)(ws + SZ);       // [b,w,t,cr]
  unsigned short* gg    = (unsigned short*)(ws + 2 * SZ);   // [b,cr,hw]
  float*          attnF = (float*)(ws + 3 * SZ);            // fp32 attn (2*SZ bytes)
  unsigned short* out1T = (unsigned short*)(ws + 3 * SZ);   // aliases attnF (dead then)
  unsigned short* attnH = (unsigned short*)(ws + 5 * SZ);   // bf16 probs
  float*          sums  = (float*)(ws + 6 * SZ);
  float*          yv    = (float*)(ws + 6 * SZ + 4096);
  unsigned short* wbf   = (unsigned short*)(ws + 6 * SZ + 8192); // 4x131072 bf16
  unsigned short* wpb = wbf;
  unsigned short* wtb = wbf + 131072;
  unsigned short* wgb = wbf + 262144;
  unsigned short* wmb = wbf + 393216;

  (void)hipMemsetAsync(sums, 0, BATCH * 4 * sizeof(float), stream);

  convw_kernel<<<dim3(256, 4), 256, 0, stream>>>(Wp, Wt, Wg, Wm, wbf);
  proj_kernel<<<dim3(32, BATCH), 256, 0, stream>>>(xa, xb, wpb, wtb, wgb, phiT, thT, gg);
  attn_kernel<<<dim3(4, BATCH * 4), 256, 0, stream>>>(phiT, thT, attnF, sums);
  se_kernel<<<1, 128, 0, stream>>>(sums, w1, w2, yv);
  softmax_kernel<<<(BATCH * 65536) / 256, 256, 0, stream>>>(attnF, yv, attnH);
  outw_kernel<<<dim3(4, BATCH * 4), 256, 0, stream>>>(gg, attnH, out1T);
  mask_kernel<<<dim3(32, 2, BATCH), 256, 0, stream>>>(wmb, out1T, xa, xb, (float*)d_out);
}
